// GINEEquivariant_58171037057266
// MI455X (gfx1250) — matrix operand order repaired and checked by reading, not compiled, not run on hardware
//
#include <hip/hip_runtime.h>
#include <hip/hip_bf16.h>

// ---------------------------------------------------------------------------
// GPS-style GNN (SPE + GINE + equivariant PE conv) for MI455X (gfx1250).
// Memory/atomic-bound: fuse all per-edge math (ea[E,64] never hits HBM),
// scatter-adds via global_atomic_add_f32 into L2-resident accumulators.
// All 16-divisible GEMMs use v_wmma_f32_16x16x32_f16 with branch-free
// ds_load_b128 fragment assembly (A row-major f16, B = transposed weights).
// ---------------------------------------------------------------------------

constexpr int kN  = 50000;
constexpr int kE  = 500000;
constexpr int kB  = 32;
constexpr int kH  = 64;
constexpr int kHP = 16;
constexpr int kP  = 8;
constexpr int kL  = 4;
constexpr float kEPS = 1e-5f;

typedef __attribute__((ext_vector_type(16))) _Float16 v16h;
typedef __attribute__((ext_vector_type(8)))  _Float16 v8h;
typedef __attribute__((ext_vector_type(8)))  float    v8f;

__device__ inline v16h cat8(v8h lo, v8h hi) {
  return __builtin_shufflevector(lo, hi, 0, 1, 2, 3, 4, 5, 6, 7, 8, 9, 10, 11,
                                 12, 13, 14, 15);
}

// A fragment (16x32 f16) from row-major f16 tile [16+ rows][ldk halves].
// Lane: m = lane&15, khalf = lane>>4; halves are two contiguous runs of 8:
//   [m][kofs + khalf*8 .. +7]  and  [m][kofs + 16 + khalf*8 .. +7]
__device__ inline v16h a_frag16(const _Float16* __restrict__ tile, int ldk,
                                int kofs, int lane) {
  int m = lane & 15, kh = (lane >> 4) * 8;
  const _Float16* r = tile + m * ldk + kofs + kh;
  return cat8(*(const v8h*)r, *(const v8h*)(r + 16));
}

// B fragment (32x16 f16) from TRANSPOSED f16 weights wt[n][ldk halves]:
// lane: n = n0 + lane&15, g = lane>>4; two runs of 8 at K = kofs+g*8 and kofs+16+g*8.
__device__ inline v16h b_frag16(const _Float16* __restrict__ wt, int ldk,
                                int kofs, int n0, int lane) {
  const _Float16* p = wt + (n0 + (lane & 15)) * ldk + kofs + (lane >> 4) * 8;
  return cat8(*(const v8h*)p, *(const v8h*)(p + 16));
}
// C/D (16x16 f32): lane L, vgpr r -> M = (L/16)*8 + r, N = L%16.

// ---------------------------------------------------------------------------
__global__ void fill_zero_kernel(float* __restrict__ p, int n) {
  for (int i = blockIdx.x * blockDim.x + threadIdx.x; i < n;
       i += gridDim.x * blockDim.x)
    p[i] = 0.f;
}

__global__ void init_state_kernel(const float* __restrict__ x,
                                  const float* __restrict__ pe,
                                  float* __restrict__ xbuf,
                                  float* __restrict__ zbuf) {
  int i = blockIdx.x * blockDim.x + threadIdx.x;
  if (i < kN * kH) xbuf[i] = x[i];
  if (i < kN * kHP * kP) {
    int n = i >> 7, p = i & 7;
    zbuf[i] = pe[n * kP + p];
  }
}

// phi = relu(Lambda @ We1[i] + be1[i])  -> [B, HP*P]
__global__ void phi_kernel(const float* __restrict__ Lambda,
                           const float* __restrict__ We1,
                           const float* __restrict__ be1,
                           float* __restrict__ phi, int layer) {
  int b = blockIdx.x, o = threadIdx.x;            // 128 threads
  const float* W = We1 + layer * (kP * kHP * kP);
  float s = be1[layer * (kHP * kP) + o];
#pragma unroll
  for (int q = 0; q < kP; ++q)
    s += Lambda[b * kP + q] * W[q * (kHP * kP) + o];
  phi[b * (kHP * kP) + o] = fmaxf(s, 0.f);
}

// ---------------------------------------------------------------------------
// Fused edge kernel: eh -> WMMA(ea) -> GINE msg atomics -> WMMA(gate) -> z msg
// 128 threads = 4 waves, 32 edges/wave.
__global__ __launch_bounds__(128) void edge_kernel(
    const float* __restrict__ xbuf, const float* __restrict__ zbuf,
    const float* __restrict__ phi, const float* __restrict__ edge_attr,
    const int* __restrict__ ei, const int* __restrict__ batch,
    const float* __restrict__ Wee, const float* __restrict__ bee,
    const float* __restrict__ Wgate, const float* __restrict__ bgate,
    float* __restrict__ agg, float* __restrict__ zagg, int layer) {
  __shared__ __align__(16) _Float16 wee_t[kH][32];   // [n][k], k>=16 zeroed
  __shared__ __align__(16) _Float16 wg_t[kHP][kH];   // [n][k]
  __shared__ __align__(16) _Float16 eh_s[4][32][32]; // rows: k>=16 zeroed
  __shared__ __align__(16) _Float16 ea_h[4][32][kH];
  __shared__ float gate_s[4][32][kHP];

  int tid = threadIdx.x, lane = tid & 31, w = tid >> 5;
  // Stage transposed f16 weights (zero-pad Wee K to 32).
  for (int i = tid; i < kH * 32; i += 128) {
    int n = i >> 5, k = i & 31;
    wee_t[n][k] = (k < kHP) ? (_Float16)Wee[layer * kHP * kH + k * kH + n]
                            : (_Float16)0.f;
  }
  for (int i = tid; i < kHP * kH; i += 128) {
    int n = i >> 6, k = i & 63;
    wg_t[n][k] = (_Float16)Wgate[layer * kH * kHP + k * kHP + n];
  }
  __syncthreads();

  int e0 = blockIdx.x * 128 + w * 32 + lane;
  bool active = e0 < kE;
  int e = active ? e0 : (kE - 1);
  int src = ei[e], dst = ei[kE + e];
  int gid = batch[src];
  const float* zs = zbuf + src * (kHP * kP);
  const float* zd = zbuf + dst * (kHP * kP);
  const float* ph = phi + gid * (kHP * kP);

  // eh[h] = sum_p phi * z_src * z_dst ; packed f16 row (upper 16 zero).
  {
    float ehv[kHP];
#pragma unroll
    for (int h = 0; h < kHP; ++h) {
      float s = 0.f;
#pragma unroll
      for (int p = 0; p < kP; ++p)
        s += ph[h * kP + p] * zs[h * kP + p] * zd[h * kP + p];
      ehv[h] = s;
    }
    v8h p0, p1, zz = {};
#pragma unroll
    for (int j = 0; j < 8; ++j) { p0[j] = (_Float16)ehv[j]; p1[j] = (_Float16)ehv[8 + j]; }
    *(v8h*)&eh_s[w][lane][0] = p0;
    *(v8h*)&eh_s[w][lane][8] = p1;
    *(v8h*)&eh_s[w][lane][16] = zz;
    *(v8h*)&eh_s[w][lane][24] = zz;
  }
  __syncthreads();

  // ea = eh @ Wee   (K = 16 zero-padded to 32)
  {
    int nl = lane & 15, mb = (lane >> 4) * 8;
#pragma unroll
    for (int t = 0; t < 2; ++t) {
      v16h af = a_frag16(&eh_s[w][t * 16][0], 32, 0, lane);
#pragma unroll
      for (int c = 0; c < 4; ++c) {
        v16h bf = b_frag16(&wee_t[0][0], 32, 0, c * 16, lane);
        v8f acc = {};
        acc = __builtin_amdgcn_wmma_f32_16x16x32_f16(false, af, false, bf,
                                                     (short)0, acc, false, false);
        int n = c * 16 + nl;
#pragma unroll
        for (int r = 0; r < 8; ++r)
          ea_h[w][t * 16 + mb + r][n] = (_Float16)acc[r];
      }
    }
  }
  __syncthreads();

  // ea += bee + edge_attr; GINE message: atomic scatter of relu(x_src + ea)
  {
    const float* eat = edge_attr + (long)e * kH;
    const float* xs = xbuf + src * kH;
    const float* be = bee + layer * kH;
    _Float16* earow = &ea_h[w][lane][0];
#pragma unroll
    for (int c = 0; c < 8; ++c) {
      v8h ev = *(const v8h*)(earow + c * 8);
      v8h ov;
#pragma unroll
      for (int j = 0; j < 8; ++j) {
        int k = c * 8 + j;
        float v = (float)ev[j] + be[k] + eat[k];
        ov[j] = (_Float16)v;
        float msg = fmaxf(xs[k] + v, 0.f);
        if (active) atomicAdd(&agg[dst * kH + k], msg);
      }
      *(v8h*)(earow + c * 8) = ov;
    }
  }
  __syncthreads();

  // gate = relu(ea @ Wg + bg)   (K = 64, two k-steps)
  {
    const float* bgv = bgate + layer * kHP;
    int nl = lane & 15, mb = (lane >> 4) * 8;
#pragma unroll
    for (int t = 0; t < 2; ++t) {
      v8f acc = {};
#pragma unroll
      for (int s = 0; s < 2; ++s) {
        v16h af = a_frag16(&ea_h[w][t * 16][0], kH, s * 32, lane);
        v16h bf = b_frag16(&wg_t[0][0], kH, s * 32, 0, lane);
        acc = __builtin_amdgcn_wmma_f32_16x16x32_f16(false, af, false, bf,
                                                     (short)0, acc, false, false);
      }
#pragma unroll
      for (int r = 0; r < 8; ++r)
        gate_s[w][t * 16 + mb + r][nl] = fmaxf(acc[r] + bgv[nl], 0.f);
    }
  }
  __syncthreads();

  // z message: atomic scatter of relu(z_src + gate)
#pragma unroll
  for (int h = 0; h < kHP; ++h) {
    float g = gate_s[w][lane][h];
#pragma unroll
    for (int p = 0; p < kP; ++p) {
      float m = fmaxf(zs[h * kP + p] + g, 0.f);
      if (active) atomicAdd(&zagg[dst * (kHP * kP) + h * kP + p], m);
    }
  }
}

// ---------------------------------------------------------------------------
// Node GEMM (N x 64 @ 64 x 64) with fused BN-stat accumulation.
// stage 0: A = x + agg,  W1, b1 -> tbuf, stats_out = raw sums of t
// stage 1: A = relu(g1*(t-mu)*rstd+beta1), W2, b2 -> h2, stats_out for outer BN
// 256 threads = 8 waves, 16 rows/wave.
__global__ __launch_bounds__(256) void gemm_kernel(
    const float* __restrict__ in0, const float* __restrict__ in1,
    const float* __restrict__ Wmat, const float* __restrict__ bias,
    const float* __restrict__ gamma, const float* __restrict__ beta,
    float* __restrict__ outbuf, const float* __restrict__ stats_in,
    float* __restrict__ stats_out, int stage) {
  __shared__ __align__(16) _Float16 w_t[kH][kH];   // [n][k] transposed
  __shared__ __align__(16) _Float16 a_s[8][16][kH];
  __shared__ float sp[2][kH];
  __shared__ float mu_s[kH], rs_s[kH];

  int tid = threadIdx.x, lane = tid & 31, w = tid >> 5;
  for (int i = tid; i < kH * kH; i += 256) {
    int n = i >> 6, k = i & 63;
    w_t[n][k] = (_Float16)Wmat[k * kH + n];
  }
  if (tid < kH) { sp[0][tid] = 0.f; sp[1][tid] = 0.f; }
  if (stage == 1 && tid < kH) {
    float mu = stats_in[tid] * (1.0f / kN);
    float var = stats_in[kH + tid] * (1.0f / kN) - mu * mu;
    mu_s[tid] = mu;
    rs_s[tid] = rsqrtf(var + kEPS);
  }
  __syncthreads();

  int rowbase = blockIdx.x * 128 + w * 16;
  // Stage A: lane handles row lane/2, half (lane&1)*32; packed f16 stores.
  {
    int r = lane >> 1, k0 = (lane & 1) * 32;
    int node = rowbase + r;
    if (node >= kN) node = kN - 1;
#pragma unroll
    for (int c = 0; c < 4; ++c) {
      v8h pk;
#pragma unroll
      for (int j = 0; j < 8; ++j) {
        int k = k0 + c * 8 + j;
        float v;
        if (stage == 0) {
          v = in0[node * kH + k] + in1[node * kH + k];
        } else {
          float t = in0[node * kH + k];
          v = fmaxf(gamma[k] * (t - mu_s[k]) * rs_s[k] + beta[k], 0.f);
        }
        pk[j] = (_Float16)v;
      }
      *(v8h*)&a_s[w][r][k0 + c * 8] = pk;
    }
  }
  __syncthreads();

  v16h a0 = a_frag16(&a_s[w][0][0], kH, 0, lane);
  v16h a1 = a_frag16(&a_s[w][0][0], kH, 32, lane);
  int nl = lane & 15, mb = (lane >> 4) * 8;
#pragma unroll
  for (int c = 0; c < 4; ++c) {
    v16h b0 = b_frag16(&w_t[0][0], kH, 0, c * 16, lane);
    v16h b1 = b_frag16(&w_t[0][0], kH, 32, c * 16, lane);
    v8f acc = {};
    acc = __builtin_amdgcn_wmma_f32_16x16x32_f16(false, a0, false, b0, (short)0,
                                                 acc, false, false);
    acc = __builtin_amdgcn_wmma_f32_16x16x32_f16(false, a1, false, b1, (short)0,
                                                 acc, false, false);
    int n = c * 16 + nl;
    float bv = bias[n];
    float psum = 0.f, psq = 0.f;
#pragma unroll
    for (int r = 0; r < 8; ++r) {
      int node = rowbase + mb + r;
      float v = acc[r] + bv;
      if (node < kN) {
        outbuf[node * kH + n] = v;
        psum += v;
        psq += v * v;
      }
    }
    atomicAdd(&sp[0][n], psum);   // ds_add_f32 (LDS), low contention
    atomicAdd(&sp[1][n], psq);
  }
  __syncthreads();
  if (tid < kH) {
    atomicAdd(&stats_out[tid], sp[0][tid]);
    atomicAdd(&stats_out[kH + tid], sp[1][tid]);
  }
}

// ---------------------------------------------------------------------------
// Finalize: x = (relu?)(BN(h2));  z = (z + zagg) einsum Wp + bp.
// 256 threads = 2 nodes (128 threads each: thread = (k,p)).
__global__ __launch_bounds__(256) void finalize_kernel(
    const float* __restrict__ h2, const float* __restrict__ stats2,
    const float* __restrict__ gn, const float* __restrict__ bnb,
    const float* __restrict__ zagg, const float* __restrict__ Wp,
    const float* __restrict__ bp, float* __restrict__ xbuf,
    float* __restrict__ zbuf, int last) {
  __shared__ float znew[2][kHP * kP];
  __shared__ float mu_s[kH], rs_s[kH];
  __shared__ float wp_s[kHP * kHP];
  int tid = threadIdx.x;
  if (tid < kH) {
    float mu = stats2[tid] * (1.0f / kN);
    float var = stats2[kH + tid] * (1.0f / kN) - mu * mu;
    mu_s[tid] = mu;
    rs_s[tid] = rsqrtf(var + kEPS);
  }
  if (tid < kHP * kHP) wp_s[tid] = Wp[tid];
  int grp = tid >> 7, t = tid & 127;
  int node = blockIdx.x * 2 + grp;
  bool valid = node < kN;
  int nc = valid ? node : (kN - 1);
  __syncthreads();
  if (t < kH) {
    float v = gn[t] * (h2[nc * kH + t] - mu_s[t]) * rs_s[t] + bnb[t];
    if (!last) v = fmaxf(v, 0.f);
    if (valid) xbuf[nc * kH + t] = v;
  }
  znew[grp][t] = zbuf[nc * (kHP * kP) + t] + zagg[nc * (kHP * kP) + t];
  __syncthreads();
  int k = t >> 3, p = t & 7;
  float s = bp[k];
#pragma unroll
  for (int h = 0; h < kHP; ++h) s += znew[grp][h * kP + p] * wp_s[h * kHP + k];
  if (valid) zbuf[nc * (kHP * kP) + t] = s;
}

__global__ void copy_out_kernel(const float* __restrict__ xbuf,
                                const float* __restrict__ zbuf,
                                float* __restrict__ out) {
  int i = blockIdx.x * blockDim.x + threadIdx.x;
  if (i < kN * kH) out[i] = xbuf[i];
  int j = i - kN * kH;
  if (j >= 0 && j < kN * kHP * kP) out[kN * kH + j] = zbuf[j];
}

// ---------------------------------------------------------------------------
extern "C" void kernel_launch(void* const* d_in, const int* in_sizes, int n_in,
                              void* d_out, int out_size, void* d_ws,
                              size_t ws_size, hipStream_t stream) {
  const float* x      = (const float*)d_in[0];
  const float* pe     = (const float*)d_in[1];
  const float* Lambda = (const float*)d_in[2];
  const float* eattr  = (const float*)d_in[3];
  const int*   ei     = (const int*)d_in[4];
  const int*   batch  = (const int*)d_in[5];
  const float* We1    = (const float*)d_in[6];
  const float* be1    = (const float*)d_in[7];
  const float* Wee    = (const float*)d_in[8];
  const float* bee    = (const float*)d_in[9];
  const float* W1     = (const float*)d_in[10];
  const float* b1     = (const float*)d_in[11];
  const float* g1     = (const float*)d_in[12];
  const float* beta1  = (const float*)d_in[13];
  const float* W2     = (const float*)d_in[14];
  const float* b2     = (const float*)d_in[15];
  const float* Wg     = (const float*)d_in[16];
  const float* bg     = (const float*)d_in[17];
  const float* Wp     = (const float*)d_in[18];
  const float* bp     = (const float*)d_in[19];
  const float* gn     = (const float*)d_in[20];
  const float* bnb    = (const float*)d_in[21];

  float* ws = (float*)d_ws;
  const int NH = kN * kH;          // 3.2M
  const int NZ = kN * kHP * kP;    // 6.4M
  float* xbuf = ws;                // NH
  float* zbuf = xbuf + NH;         // NZ
  float* agg  = zbuf + NZ;         // NH
  float* zagg = agg + NH;          // NZ
  float* stats = zagg + NZ;        // 256 (sum1,sq1 | sum2,sq2)
  float* phi  = stats + 256;       // B*HP*P = 4096
  float* tbuf = phi + kB * kHP * kP;  // NH

  init_state_kernel<<<(NZ + 255) / 256, 256, 0, stream>>>(x, pe, xbuf, zbuf);

  for (int i = 0; i < kL; ++i) {
    // zero agg, zagg, stats (contiguous region)
    fill_zero_kernel<<<2048, 256, 0, stream>>>(agg, NH + NZ + 256);
    phi_kernel<<<kB, kHP * kP, 0, stream>>>(Lambda, We1, be1, phi, i);
    edge_kernel<<<(kE + 127) / 128, 128, 0, stream>>>(
        xbuf, zbuf, phi, eattr, ei, batch, Wee, bee, Wg, bg, agg, zagg, i);
    gemm_kernel<<<(kN + 127) / 128, 256, 0, stream>>>(
        xbuf, agg, W1 + i * kH * kH, b1 + i * kH, nullptr, nullptr, tbuf,
        nullptr, stats, 0);
    gemm_kernel<<<(kN + 127) / 128, 256, 0, stream>>>(
        tbuf, nullptr, W2 + i * kH * kH, b2 + i * kH, g1 + i * kH,
        beta1 + i * kH, agg, stats, stats + 128, 1);
    finalize_kernel<<<(kN + 1) / 2, 256, 0, stream>>>(
        agg, stats + 128, gn + i * kH, bnb + i * kH, zagg, Wp + i * kHP * kHP,
        bp + i * kHP, xbuf, zbuf, (i == kL - 1) ? 1 : 0);
  }

  copy_out_kernel<<<(NH + NZ + 255) / 256, 256, 0, stream>>>(xbuf, zbuf,
                                                             (float*)d_out);
}
